// SCEncoder_44521630990669
// MI455X (gfx1250) — matrix operand order repaired and checked by reading, not compiled, not run on hardware
//
#include <hip/hip_runtime.h>
#include <hip/hip_bf16.h>
#include <math.h>

// ---------------------------------------------------------------------------
// Performer (FAVOR+) encoder forward for MI455X (gfx1250, wave32, WMMA).
// All large matmuls use v_wmma_f32_16x16x32_f16 (f16 inputs, f32 accum).
// Main GEMM: 32x64 register tile per wave (8 WMMAs / K-step, 3x operand reuse).
// ---------------------------------------------------------------------------

typedef _Float16 v8h  __attribute__((ext_vector_type(8)));
typedef _Float16 v16h __attribute__((ext_vector_type(16)));
typedef float    v8f  __attribute__((ext_vector_type(8)));

#define E_GENES 16906
#define TKK     2048
#define D_MODEL 256
#define N_HEADS 8
#define D_HEAD  32
#define M_FEAT  110
#define M_PAD   112
#define D_FF    1024
#define N_TOK   2049           // CLS + 2048 tokens
#define N_PADT  2080           // padded tokens per batch (mult of 32)
#define B_SZ    8
#define BH_SZ   64             // B * H
#define R_ROWS  (B_SZ * N_PADT)

#define DN_SCALE 0.42044820762685725f   // 32^-0.25
#define DN2      0.17677669529663687f   // 32^-0.5
#define RATIO_M  0.09534625892455922f   // 110^-0.5

#define WMMA_F16(a, b, c) __builtin_amdgcn_wmma_f32_16x16x32_f16( \
    false, (a), false, (b), (short)0, (c), false, false)

#define SHUF16(lo, hi) __builtin_shufflevector((lo), (hi), \
    0, 1, 2, 3, 4, 5, 6, 7, 8, 9, 10, 11, 12, 13, 14, 15)

// ------------------------- weight prep --------------------------------------

// W: [L][K][N] f32  ->  Wt: [L][N][K] f16   (B-operand layout for WMMA)
__global__ void transpose_h_kernel(const float* __restrict__ W,
                                   _Float16* __restrict__ Wt, int K, int N) {
    size_t per = (size_t)K * N;
    size_t idx = (size_t)blockIdx.x * 256 + threadIdx.x;
    if (idx >= per) return;
    size_t layer = blockIdx.y;
    int k = (int)(idx / N), n = (int)(idx % N);
    Wt[layer * per + (size_t)n * K + k] = (_Float16)W[layer * per + idx];
}

// proj: [L][110][32] f32 -> projh: [L][112][32] f16 (zero-padded rows)
__global__ void proj_prep_kernel(const float* __restrict__ P,
                                 _Float16* __restrict__ Ph) {
    int layer = blockIdx.y;
    int idx = blockIdx.x * 256 + threadIdx.x;   // < 112*32
    if (idx >= M_PAD * D_HEAD) return;
    int m = idx / D_HEAD, d = idx % D_HEAD;
    Ph[(size_t)layer * M_PAD * D_HEAD + idx] =
        (m < M_FEAT) ? (_Float16)P[((size_t)layer * M_FEAT + m) * D_HEAD + d]
                     : (_Float16)0.0f;
}

// ------------------------- top-k + tokenize ---------------------------------

__device__ __forceinline__ int tok_of(float v, float sumc, float emax) {
    float e = log1pf(v / sumc * 1.0e4f);
    float step = emax / 7.0f;
    int cnt = 0;
#pragma unroll
    for (int j = 0; j < 7; ++j) if (step * (float)j < e) cnt++;   // searchsorted 'left'
    return cnt > 6 ? 6 : cnt;
}

__global__ __launch_bounds__(256)
void topk_tokens_kernel(const float* __restrict__ expr, int* __restrict__ tokens) {
    const int b = blockIdx.x, t = threadIdx.x;
    const float* row = expr + (size_t)b * E_GENES;
    __shared__ int   sc[256];
    __shared__ float sf[256];

    // exact k-th largest via binary search on float bit pattern (all vals >= 0)
    unsigned lo = 0u, hi = 0xFFFFFFFFu;
    for (int it = 0; it < 32; ++it) {
        unsigned mid = (unsigned)(((unsigned long long)lo + hi + 1ull) >> 1);
        int c = 0;
        for (int i = t; i < E_GENES; i += 256)
            if (__float_as_uint(row[i]) >= mid) c++;
        sc[t] = c; __syncthreads();
        for (int s = 128; s > 0; s >>= 1) { if (t < s) sc[t] += sc[t + s]; __syncthreads(); }
        int cnt = sc[0]; __syncthreads();
        if (cnt >= TKK) lo = mid; else hi = mid - 1;
    }
    const unsigned T = lo;
    const float Tval = __uint_as_float(T);

    // count strictly-greater, their sum, and global max
    int c = 0; float s = 0.0f, mx = -3.4e38f;
    for (int i = t; i < E_GENES; i += 256) {
        float v = row[i]; unsigned u = __float_as_uint(v);
        if (u > T) { c++; s += v; }
        mx = fmaxf(mx, v);
    }
    sc[t] = c; sf[t] = s; __syncthreads();
    for (int st = 128; st > 0; st >>= 1) {
        if (t < st) { sc[t] += sc[t + st]; sf[t] += sf[t + st]; } __syncthreads();
    }
    int cntGT = sc[0]; float sumGT = sf[0]; __syncthreads();
    sf[t] = mx; __syncthreads();
    for (int st = 128; st > 0; st >>= 1) { if (t < st) sf[t] = fmaxf(sf[t], sf[t + st]); __syncthreads(); }
    float maxv = sf[0]; __syncthreads();

    int   ties = TKK - cntGT;
    float total = sumGT + (float)ties * Tval;
    float sumc  = fmaxf(total, 1e-12f);
    float emax  = log1pf(maxv / sumc * 1.0e4f);

    // deterministic compaction: per-thread contiguous chunk + shared prefix scan
    const int CH = (E_GENES + 255) / 256;
    int begin = t * CH, end = min(begin + CH, E_GENES);
    int mine = 0;
    for (int i = begin; i < end; ++i) if (__float_as_uint(row[i]) > T) mine++;
    sc[t] = mine; __syncthreads();
    for (int st = 1; st < 256; st <<= 1) {
        int add = (t >= st) ? sc[t - st] : 0;
        __syncthreads();
        sc[t] += add;
        __syncthreads();
    }
    int pos = sc[t] - mine;
    int* trow = tokens + (size_t)b * TKK;
    for (int i = begin; i < end; ++i) {
        float v = row[i];
        if (__float_as_uint(v) > T) trow[pos++] = tok_of(v, sumc, emax);
    }
    __syncthreads();
    int totGT = sc[255];
    int tieTok = tok_of(Tval, sumc, emax);
    for (int p = totGT + t; p < TKK; p += 256) trow[p] = tieTok;
}

__global__ __launch_bounds__(256)
void build_x_kernel(const float* __restrict__ emb, const float* __restrict__ cls,
                    const int* __restrict__ tokens, float* __restrict__ X) {
    int row = blockIdx.x, t = threadIdx.x;
    int b = row / N_PADT, n = row % N_PADT;
    float v;
    if (n == 0)            v = cls[t];
    else if (n < N_TOK)    v = emb[(size_t)tokens[(size_t)b * TKK + n - 1] * D_MODEL + t];
    else                   v = 0.0f;
    X[(size_t)row * D_MODEL + t] = v;
}

// ------------------------- layernorm ----------------------------------------

__global__ __launch_bounds__(256)
void ln_kernel(const float* __restrict__ X, const float* __restrict__ g,
               const float* __restrict__ bta, _Float16* __restrict__ H) {
    int row = blockIdx.x, t = threadIdx.x;
    __shared__ float sm[256];
    float v = X[(size_t)row * D_MODEL + t];
    sm[t] = v; __syncthreads();
    for (int s = 128; s > 0; s >>= 1) { if (t < s) sm[t] += sm[t + s]; __syncthreads(); }
    float mu = sm[0] * (1.0f / D_MODEL); __syncthreads();
    float d = v - mu;
    sm[t] = d * d; __syncthreads();
    for (int s = 128; s > 0; s >>= 1) { if (t < s) sm[t] += sm[t + s]; __syncthreads(); }
    float var = sm[0] * (1.0f / D_MODEL);
    float o = d * rsqrtf(var + 1e-5f) * g[t] + bta[t];
    H[(size_t)row * D_MODEL + t] = (_Float16)o;
}

// ------------------------- generic WMMA GEMM --------------------------------
// C[M x N] = A[M x K] @ B[K x N] (+bias)(+gelu)(+residual)
// A: f16 row-major. Bt: f16 TRANSPOSED ([N][K] row-major) so each lane's
// B-fragment is one contiguous 32B v16h load.
// Register blocking: each wave computes a 32x64 tile (2 A-frags x 4 B-frags =
// 8 WMMAs per K-step). 4 waves/block arranged 2x2 -> 64x128 macrotile.

__global__ __launch_bounds__(128)
void gemm_wmma_kernel(const _Float16* __restrict__ A, const _Float16* __restrict__ Bt,
                      const float* __restrict__ bias, const float* __restrict__ resid,
                      float* __restrict__ Cf, _Float16* __restrict__ Ch, float hscale,
                      int Ndim, int Kdim, int act) {
    const int lane  = threadIdx.x & 31;
    const int wave  = threadIdx.x >> 5;
    const int mbase = blockIdx.x * 64 + (wave & 1) * 32;
    const int nbase = blockIdx.y * 128 + (wave >> 1) * 64;
    const int kq    = 8 * (lane >> 4);

    const _Float16* ap0 = A + (size_t)(mbase + (lane & 15)) * Kdim + 2 * kq;
    const _Float16* ap1 = ap0 + (size_t)16 * Kdim;
    const _Float16* bp  = Bt + (size_t)(nbase + (lane & 15)) * Kdim + 2 * kq;
    const size_t bstride = (size_t)16 * Kdim;

    v8f acc[2][4] = {};
    for (int k0 = 0; k0 < Kdim; k0 += 32) {
        v8h a0lo = *(const v8h*)(ap0 + k0);
        v8h a0hi = *(const v8h*)(ap0 + k0 + 8);
        v8h a1lo = *(const v8h*)(ap1 + k0);
        v8h a1hi = *(const v8h*)(ap1 + k0 + 8);
        v16h af0 = SHUF16(a0lo, a0hi);
        v16h af1 = SHUF16(a1lo, a1hi);
#pragma unroll
        for (int j = 0; j < 4; ++j) {
            v16h bf = *(const v16h*)(bp + (size_t)j * bstride + k0);
            acc[0][j] = WMMA_F16(af0, bf, acc[0][j]);
            acc[1][j] = WMMA_F16(af1, bf, acc[1][j]);
        }
    }

#pragma unroll
    for (int mi = 0; mi < 2; ++mi) {
#pragma unroll
        for (int j = 0; j < 4; ++j) {
            int col = nbase + j * 16 + (lane & 15);
            float bv = bias ? bias[col] : 0.0f;
#pragma unroll
            for (int r = 0; r < 8; ++r) {
                int mr = mbase + mi * 16 + r + 8 * (lane >> 4);
                float v = acc[mi][j][r] + bv;
                if (act == 1) v = 0.5f * v * (1.0f + erff(v * 0.7071067811865475f)); // exact GELU
                size_t idx = (size_t)mr * Ndim + col;
                if (resid) v += resid[idx];
                if (Cf) Cf[idx] = v;
                if (Ch) Ch[idx] = (_Float16)(v * hscale);
            }
        }
    }
}

// ------------------------- FAVOR+ feature GEMM ------------------------------
// xd[bh][n][m] = (q*dn)[n, d] @ projT[d, m]   (K = 32, one WMMA per tile)

__global__ __launch_bounds__(32)
void feat_wmma_kernel(const _Float16* __restrict__ Qh, const _Float16* __restrict__ Ph,
                      float* __restrict__ Xd) {
    int bh = blockIdx.z; int b = bh >> 3, h = bh & 7;
    int nbase = blockIdx.x * 16;
    int mbase = blockIdx.y * 16;
    int lane = threadIdx.x & 31;

    int row = b * N_PADT + nbase + (lane & 15);
    const _Float16* ap = Qh + (size_t)row * D_MODEL + h * D_HEAD + 16 * (lane >> 4);
    v8h alo = *(const v8h*)ap;
    v8h ahi = *(const v8h*)(ap + 8);
    v16h af = SHUF16(alo, ahi);

    int col = mbase + (lane & 15);
    v16h bf = *(const v16h*)(Ph + (size_t)col * D_HEAD + 16 * (lane >> 4));

    v8f acc = {};
    acc = WMMA_F16(af, bf, acc);

#pragma unroll
    for (int r = 0; r < 8; ++r) {
        int n = nbase + r + 8 * (lane >> 4);
        Xd[((size_t)bh * N_PADT + n) * M_PAD + col] = acc[r];
    }
}

// ------------------------- key global max (2-stage) -------------------------

__global__ __launch_bounds__(256)
void kmax1_kernel(const float* __restrict__ Xd, float* __restrict__ part) {
    int bh = blockIdx.x, t = threadIdx.x;
    __shared__ float sm[256];
    const float* base = Xd + (size_t)bh * N_PADT * M_PAD;
    float m = -3.4e38f;
    for (int idx = t; idx < N_TOK * M_FEAT; idx += 256) {
        int n = idx / M_FEAT, mm = idx % M_FEAT;
        m = fmaxf(m, base[(size_t)n * M_PAD + mm]);
    }
    sm[t] = m; __syncthreads();
    for (int s = 128; s > 0; s >>= 1) { if (t < s) sm[t] = fmaxf(sm[t], sm[t + s]); __syncthreads(); }
    if (t == 0) part[bh] = sm[0];
}

__global__ __launch_bounds__(64)
void kmax2_kernel(const float* __restrict__ part, float* __restrict__ stab) {
    int t = threadIdx.x;
    __shared__ float sm[64];
    sm[t] = part[t]; __syncthreads();
    for (int s = 32; s > 0; s >>= 1) { if (t < s) sm[t] = fmaxf(sm[t], sm[t + s]); __syncthreads(); }
    if (t == 0) stab[0] = sm[0];
}

// ------------------------- feature finalize (in place) ----------------------
// qp/kp = ratio * (exp(xd - diag - stab) + 1e-3); key padding rows/cols -> 0

__global__ __launch_bounds__(128)
void finalize_feat_kernel(float* __restrict__ Xd, const float* __restrict__ Qf,
                          const float* __restrict__ stabp, int isQ) {
    int n = blockIdx.x, bh = blockIdx.y;
    int b = bh >> 3, h = bh & 7;
    int t = threadIdx.x;
    float* rowp = Xd + ((size_t)bh * N_PADT + n) * M_PAD;
    __shared__ float sm[128];
    __shared__ float sdiag, smax;

    float v = (t < M_FEAT) ? rowp[t] : -3.4e38f;
    sm[t] = v; __syncthreads();
    for (int s = 64; s > 0; s >>= 1) { if (t < s) sm[t] = fmaxf(sm[t], sm[t + s]); __syncthreads(); }
    if (t == 0) {
        const float* q = Qf + ((size_t)(b * N_PADT + n)) * D_MODEL + h * D_HEAD;
        float diag = 0.0f;
        for (int d = 0; d < D_HEAD; ++d) diag += q[d] * q[d];
        sdiag = 0.5f * diag * DN2;
        smax  = isQ ? sm[0] : stabp[0];
    }
    __syncthreads();

    float out;
    if (!isQ && n >= N_TOK)  out = 0.0f;                 // padded key rows must not pollute sums
    else if (t >= M_FEAT)    out = 0.0f;                 // padded feature cols
    else                     out = RATIO_M * (expf(v - sdiag - smax) + 1e-3f);
    if (t < M_PAD) rowp[t] = out;
}

// ------------------------- k_sum --------------------------------------------

__global__ __launch_bounds__(128)
void ksum_kernel(const float* __restrict__ Kp, float* __restrict__ Ksum) {
    int bh = blockIdx.x, t = threadIdx.x;
    if (t >= M_PAD) return;
    const float* base = Kp + (size_t)bh * N_PADT * M_PAD + t;
    float s = 0.0f;
    for (int n = 0; n < N_TOK; ++n) s += base[(size_t)n * M_PAD];
    Ksum[(size_t)bh * M_PAD + t] = s;
}

// ------------------------- context GEMM: ctx = kp^T @ v_head ----------------
// A = kp^T [m x n] (gathered, strided), B = v_head [n x d] (gathered), WMMA K-loop over n.

__global__ __launch_bounds__(32)
void ctx_wmma_kernel(const float* __restrict__ Kp, const float* __restrict__ Vf,
                     float* __restrict__ Ctx) {
    int bh = blockIdx.z; int b = bh >> 3, h = bh & 7;
    int mbase = blockIdx.x * 16;
    int dbase = blockIdx.y * 16;
    int lane = threadIdx.x & 31;
    int mrow = mbase + (lane & 15);
    int dcol = dbase + (lane & 15);
    const float* kp = Kp + (size_t)bh * N_PADT * M_PAD;
    const float* vb = Vf + (size_t)b * N_PADT * D_MODEL + h * D_HEAD;

    v8f acc = {};
    for (int n0 = 0; n0 < N_PADT; n0 += 32) {
        v16h af, bf;
#pragma unroll
        for (int j = 0; j < 16; ++j) {
            int na = n0 + 16 * (lane >> 4) + j;   // A K-index
            af[j] = (_Float16)kp[(size_t)na * M_PAD + mrow];
            int nb = n0 + 16 * (lane >> 4) + j;                        // B K-index
            bf[j] = (_Float16)vb[(size_t)nb * D_MODEL + dcol];
        }
        acc = WMMA_F16(af, bf, acc);
    }
#pragma unroll
    for (int r = 0; r < 8; ++r) {
        int m = mbase + r + 8 * (lane >> 4);
        Ctx[((size_t)bh * M_PAD + m) * D_HEAD + dcol] = acc[r];
    }
}

// ------------------------- attention combine --------------------------------
// o[n,d] = (sum_m qp[n,m]*ctx[m,d]) / (sum_m qp[n,m]*ksum[m]) -> oh f16 row layout

__global__ __launch_bounds__(32)
void attn_out_kernel(const float* __restrict__ Qp, const float* __restrict__ Ksum,
                     const float* __restrict__ Ctx, _Float16* __restrict__ Oh) {
    int n = blockIdx.x, bh = blockIdx.y;
    int b = bh >> 3, h = bh & 7;
    int lane = threadIdx.x;
    const float* qp = Qp + ((size_t)bh * N_PADT + n) * M_PAD;
    const float* ks = Ksum + (size_t)bh * M_PAD;
    const float* cx = Ctx + (size_t)bh * M_PAD * D_HEAD;
    float denom = 0.0f, acc = 0.0f;
    for (int m = 0; m < M_FEAT; ++m) {
        float q = qp[m];
        denom += q * ks[m];
        acc   += q * cx[(size_t)m * D_HEAD + lane];
    }
    Oh[((size_t)(b * N_PADT + n)) * D_MODEL + h * D_HEAD + lane] = (_Float16)(acc / denom);
}

// ------------------------- final head ---------------------------------------

__global__ __launch_bounds__(256)
void head_kernel(const float* __restrict__ X, const float* __restrict__ Wp,
                 const float* __restrict__ bp, float* __restrict__ out) {
    int b = blockIdx.x, t = threadIdx.x;
    const float* xr = X + (size_t)b * N_PADT * D_MODEL;   // row 0 of batch b (CLS)
    float acc = bp[t];
    for (int j = 0; j < D_MODEL; ++j) acc += xr[j] * Wp[(size_t)j * D_MODEL + t];
    out[(size_t)b * D_MODEL + t] = acc;
}

// ===========================================================================

extern "C" void kernel_launch(void* const* d_in, const int* in_sizes, int n_in,
                              void* d_out, int out_size, void* d_ws, size_t ws_size,
                              hipStream_t stream) {
    (void)in_sizes; (void)n_in; (void)out_size; (void)ws_size;

    const float* expr  = (const float*)d_in[0];
    const float* temb  = (const float*)d_in[1];
    const float* cls   = (const float*)d_in[2];
    const float* ln1g  = (const float*)d_in[3];
    const float* ln1b  = (const float*)d_in[4];
    const float* Wq    = (const float*)d_in[5];
    const float* bq    = (const float*)d_in[6];
    const float* Wk    = (const float*)d_in[7];
    const float* bk    = (const float*)d_in[8];
    const float* Wv    = (const float*)d_in[9];
    const float* bv    = (const float*)d_in[10];
    const float* Wo    = (const float*)d_in[11];
    const float* bo    = (const float*)d_in[12];
    const float* ln2g  = (const float*)d_in[13];
    const float* ln2b  = (const float*)d_in[14];
    const float* W1    = (const float*)d_in[15];
    const float* b1    = (const float*)d_in[16];
    const float* W2    = (const float*)d_in[17];
    const float* b2    = (const float*)d_in[18];
    const float* proj  = (const float*)d_in[19];
    const float* Wp    = (const float*)d_in[20];
    const float* bp    = (const float*)d_in[21];
    float* out = (float*)d_out;

    // ---- workspace carve-up (deterministic offsets) ----
    char* W = (char*)d_ws;
    size_t off = 0;
    auto alloc = [&](size_t bytes) -> char* {
        char* p = W + off;
        off = (off + bytes + 255) & ~(size_t)255;
        return p;
    };
    float*    x    = (float*)   alloc((size_t)R_ROWS * D_MODEL * 4);
    _Float16* hh   = (_Float16*)alloc((size_t)R_ROWS * D_MODEL * 2);
    float*    qf   = (float*)   alloc((size_t)R_ROWS * D_MODEL * 4);
    float*    kf   = (float*)   alloc((size_t)R_ROWS * D_MODEL * 4);
    float*    vf   = (float*)   alloc((size_t)R_ROWS * D_MODEL * 4);
    _Float16* qh   = (_Float16*)alloc((size_t)R_ROWS * D_MODEL * 2);
    _Float16* kh   = (_Float16*)alloc((size_t)R_ROWS * D_MODEL * 2);
    float*    fq   = (float*)   alloc((size_t)BH_SZ * N_PADT * M_PAD * 4);
    float*    fk   = (float*)   alloc((size_t)BH_SZ * N_PADT * M_PAD * 4);
    float*    ksum = (float*)   alloc((size_t)BH_SZ * M_PAD * 4);
    float*    ctx  = (float*)   alloc((size_t)BH_SZ * M_PAD * D_HEAD * 4);
    _Float16* oh   = (_Float16*)alloc((size_t)R_ROWS * D_MODEL * 2);
    _Float16* midh = (_Float16*)alloc((size_t)R_ROWS * D_FF * 2);
    float*    part = (float*)   alloc((size_t)BH_SZ * 4);
    float*    kstb = (float*)   alloc(4);
    int*      toks = (int*)     alloc((size_t)B_SZ * TKK * 4);
    _Float16* Wqt  = (_Float16*)alloc((size_t)6 * D_MODEL * D_MODEL * 2);
    _Float16* Wkt  = (_Float16*)alloc((size_t)6 * D_MODEL * D_MODEL * 2);
    _Float16* Wvt  = (_Float16*)alloc((size_t)6 * D_MODEL * D_MODEL * 2);
    _Float16* Wot  = (_Float16*)alloc((size_t)6 * D_MODEL * D_MODEL * 2);
    _Float16* W1t  = (_Float16*)alloc((size_t)6 * D_MODEL * D_FF * 2);
    _Float16* W2t  = (_Float16*)alloc((size_t)6 * D_FF * D_MODEL * 2);
    _Float16* prjh = (_Float16*)alloc((size_t)6 * M_PAD * D_HEAD * 2);

    const size_t WSQ = (size_t)D_MODEL * D_MODEL;   // 65536
    const size_t WSF = (size_t)D_MODEL * D_FF;      // 262144
    const size_t PSZ = (size_t)M_PAD * D_HEAD;      // 3584

    // ---- weight prep (f16, B-transposed) ----
    transpose_h_kernel<<<dim3((WSQ + 255) / 256, 6), 256, 0, stream>>>(Wq, Wqt, D_MODEL, D_MODEL);
    transpose_h_kernel<<<dim3((WSQ + 255) / 256, 6), 256, 0, stream>>>(Wk, Wkt, D_MODEL, D_MODEL);
    transpose_h_kernel<<<dim3((WSQ + 255) / 256, 6), 256, 0, stream>>>(Wv, Wvt, D_MODEL, D_MODEL);
    transpose_h_kernel<<<dim3((WSQ + 255) / 256, 6), 256, 0, stream>>>(Wo, Wot, D_MODEL, D_MODEL);
    transpose_h_kernel<<<dim3((WSF + 255) / 256, 6), 256, 0, stream>>>(W1, W1t, D_MODEL, D_FF);
    transpose_h_kernel<<<dim3((WSF + 255) / 256, 6), 256, 0, stream>>>(W2, W2t, D_FF, D_MODEL);
    proj_prep_kernel<<<dim3((PSZ + 255) / 256, 6), 256, 0, stream>>>(proj, prjh);

    // ---- tokenize + embed ----
    topk_tokens_kernel<<<B_SZ, 256, 0, stream>>>(expr, toks);
    build_x_kernel<<<R_ROWS, 256, 0, stream>>>(temb, cls, toks, x);

    const int MT = R_ROWS / 64;       // 260 macrotiles (64 rows each)
    dim3 gemmBlk(128);

    for (int i = 0; i < 6; ++i) {
        // ---- attention block ----
        ln_kernel<<<R_ROWS, 256, 0, stream>>>(x, ln1g + i * D_MODEL, ln1b + i * D_MODEL, hh);

        gemm_wmma_kernel<<<dim3(MT, D_MODEL / 128), gemmBlk, 0, stream>>>(
            hh, Wqt + i * WSQ, bq + i * D_MODEL, nullptr, qf, qh, DN_SCALE, D_MODEL, D_MODEL, 0);
        gemm_wmma_kernel<<<dim3(MT, D_MODEL / 128), gemmBlk, 0, stream>>>(
            hh, Wkt + i * WSQ, bk + i * D_MODEL, nullptr, kf, kh, DN_SCALE, D_MODEL, D_MODEL, 0);
        gemm_wmma_kernel<<<dim3(MT, D_MODEL / 128), gemmBlk, 0, stream>>>(
            hh, Wvt + i * WSQ, bv + i * D_MODEL, nullptr, vf, nullptr, 1.0f, D_MODEL, D_MODEL, 0);

        feat_wmma_kernel<<<dim3(N_PADT / 16, M_PAD / 16, BH_SZ), 32, 0, stream>>>(qh, prjh + i * PSZ, fq);
        feat_wmma_kernel<<<dim3(N_PADT / 16, M_PAD / 16, BH_SZ), 32, 0, stream>>>(kh, prjh + i * PSZ, fk);

        kmax1_kernel<<<BH_SZ, 256, 0, stream>>>(fk, part);
        kmax2_kernel<<<1, 64, 0, stream>>>(part, kstb);

        finalize_feat_kernel<<<dim3(N_PADT, BH_SZ), 128, 0, stream>>>(fq, qf, kstb, 1);
        finalize_feat_kernel<<<dim3(N_PADT, BH_SZ), 128, 0, stream>>>(fk, kf, kstb, 0);

        ksum_kernel<<<BH_SZ, 128, 0, stream>>>(fk, ksum);
        ctx_wmma_kernel<<<dim3(M_PAD / 16, D_HEAD / 16, BH_SZ), 32, 0, stream>>>(fk, vf, ctx);
        attn_out_kernel<<<dim3(N_PADT, BH_SZ), 32, 0, stream>>>(fq, ksum, ctx, oh);

        gemm_wmma_kernel<<<dim3(MT, D_MODEL / 128), gemmBlk, 0, stream>>>(
            oh, Wot + i * WSQ, bo + i * D_MODEL, x, x, nullptr, 1.0f, D_MODEL, D_MODEL, 0);

        // ---- FFN block ----
        ln_kernel<<<R_ROWS, 256, 0, stream>>>(x, ln2g + i * D_MODEL, ln2b + i * D_MODEL, hh);
        gemm_wmma_kernel<<<dim3(MT, D_FF / 128), gemmBlk, 0, stream>>>(
            hh, W1t + i * WSF, b1 + i * D_FF, nullptr, nullptr, midh, 1.0f, D_FF, D_MODEL, 1);
        gemm_wmma_kernel<<<dim3(MT, D_MODEL / 128), gemmBlk, 0, stream>>>(
            midh, W2t + i * WSF, b2 + i * D_MODEL, x, x, nullptr, 1.0f, D_MODEL, D_FF, 0);
    }

    head_kernel<<<B_SZ, 256, 0, stream>>>(x, Wp, bp, out);
}
